// RandomizedEnsembleQNetwork_87230785782022
// MI455X (gfx1250) — compile-verified, matrix operand order checked by reading, not run on hardware
//
#include <hip/hip_runtime.h>

typedef __bf16 bf16x16 __attribute__((ext_vector_type(16)));
typedef float  f32x8   __attribute__((ext_vector_type(8)));
typedef unsigned short u16x16 __attribute__((ext_vector_type(16)));
typedef unsigned short u16x8  __attribute__((ext_vector_type(8)));

#define B_TOT 32768
#define OBS   17
#define ACTD  6
#define INDIM 23
#define HDIM  256
#define NQ    10

#define BT   128   // batch rows per workgroup
#define KP   32    // layer-1 K padded 23 -> 32
#define S32  40    // padded row pitch (elems) for 32-wide tiles (80B: 16B-aligned, bank-safe)
#define SH   264   // padded row pitch (elems) for 256-wide tiles (528B: 16B-aligned, bank-safe)

// ---- workspace layout (bf16 elements) ----
#define WS_X_ELEMS  (B_TOT * KP)            // x concat, K-padded:   1,048,576
#define WS_W1_ELEMS (NQ * HDIM * KP)        // W1^T, K-padded:          81,920
#define WS_W2_ELEMS (NQ * HDIM * HDIM)      // W2^T:                   655,360
#define WS_W1_BASE  (WS_X_ELEMS)
#define WS_W2_BASE  (WS_X_ELEMS + WS_W1_ELEMS)

__device__ __forceinline__ unsigned short f2bf(float f) {   // prep kernel only
    union { float f; unsigned u; } v; v.f = f;
    unsigned r = v.u + 0x7FFFu + ((v.u >> 16) & 1u);
    return (unsigned short)(r >> 16);
}
__device__ __forceinline__ float bf2f(unsigned short h) {
    union { unsigned u; float f; } v; v.u = ((unsigned)h) << 16;
    return v.f;
}
__device__ __forceinline__ unsigned short f2bf_hw(float f) { // hot path: native cvt
    __bf16 b = (__bf16)f;
    return __builtin_bit_cast(unsigned short, b);
}

// One async 16B DMA: global (saddr + 32-bit voffset) -> LDS byte address.
__device__ __forceinline__ void async_b128(unsigned lds_addr, unsigned goff,
                                           const unsigned short* gbase) {
    asm volatile("global_load_async_to_lds_b128 %0, %1, %2"
                 :: "v"(lds_addr), "v"(goff), "s"(gbase) : "memory");
}

// Load one 16x32 bf16 operand tile in WMMA A-layout (B uses transposed storage).
// Lanes 0-15: K {0..7,16..23}; lanes 16-31: K {8..15,24..31}; two ds_load_b128.
__device__ __forceinline__ bf16x16 load_frag(const unsigned short* base, int stride, int lane) {
    int m = lane & 15;
    int g = (lane >> 4) & 1;
    const unsigned short* p = base + m * stride;
    u16x8 lo = *(const u16x8*)(p + 8 * g);
    u16x8 hi = *(const u16x8*)(p + 16 + 8 * g);
    u16x16 f;
#pragma unroll
    for (int i = 0; i < 8; ++i) { f[i] = lo[i]; f[8 + i] = hi[i]; }
    return __builtin_bit_cast(bf16x16, f);
}

// ---- one-shot fp32 -> bf16 repack (off critical path; ~1.8M elements) ----
__global__ void prep_bf16(const float* __restrict__ state, const float* __restrict__ action,
                          const float* __restrict__ W1, const float* __restrict__ W2,
                          unsigned short* __restrict__ ws) {
    int i = blockIdx.x * 256 + threadIdx.x;
    if (i < WS_X_ELEMS) {                                  // x = [state|action|0], K->32
        int r = i >> 5, k = i & 31;
        float v = (k < OBS)   ? state [r * OBS  + k]
                : (k < INDIM) ? action[r * ACTD + (k - OBS)] : 0.f;
        ws[i] = f2bf(v);
    } else if (i < WS_X_ELEMS + WS_W1_ELEMS) {             // W1^T [n][c][k], K->32
        int j = i - WS_X_ELEMS;
        int n = j >> 13, rem = j & 8191;
        int c = rem >> 5, k = rem & 31;
        float v = (k < INDIM) ? W1[((size_t)n * INDIM + k) * HDIM + c] : 0.f;
        ws[WS_W1_BASE + j] = f2bf(v);
    } else if (i < WS_X_ELEMS + WS_W1_ELEMS + WS_W2_ELEMS) { // W2^T [n][c][k]
        int j = i - WS_X_ELEMS - WS_W1_ELEMS;
        int n = j >> 16, rem = j & 65535;
        int c = rem >> 8, k = rem & 255;
        ws[WS_W2_BASE + j] = f2bf(W2[((size_t)n * HDIM + k) * HDIM + c]);
    }
}

__global__ __launch_bounds__(256, 1)
void ensemble_q_fused(const unsigned short* __restrict__ ws,
                      const float* __restrict__ b1, const float* __restrict__ b2,
                      const float* __restrict__ W3, const float* __restrict__ b3,
                      float* __restrict__ out) {
    // ~297 KB LDS: W2 + both activation tiles resident (CDNA5: 320 KB/WGP).
    __shared__ __align__(16) unsigned short sX  [BT   * S32];
    __shared__ __align__(16) unsigned short sW1t[HDIM * S32];
    __shared__ __align__(16) unsigned short sW2t[HDIM * SH ];
    __shared__ __align__(16) unsigned short sH1 [BT   * SH ];
    __shared__ __align__(16) unsigned short sH2 [BT   * SH ];
    __shared__ float sB1[HDIM], sB2[HDIM], sW3v[HDIM];

    const int tid  = threadIdx.x;
    const int lane = tid & 31;
    const int wave = tid >> 5;
    const int n    = blockIdx.y;
    const int rowBase = blockIdx.x * BT;

    const unsigned short* gx  = ws + (size_t)rowBase * KP;
    const unsigned short* gw1 = ws + WS_W1_BASE + (size_t)n * HDIM * KP;
    const unsigned short* gw2 = ws + WS_W2_BASE + (size_t)n * HDIM * HDIM;

    const unsigned sx_b  = (unsigned)(uintptr_t)&sX[0];    // flat addr[31:0] == LDS offset
    const unsigned sw1_b = (unsigned)(uintptr_t)&sW1t[0];
    const unsigned sw2_b = (unsigned)(uintptr_t)&sW2t[0];

    // biases / W3 via normal loads (overlap with DMA below)
    if (tid < HDIM) {
        sB1[tid]  = b1[n * HDIM + tid];
        sB2[tid]  = b2[n * HDIM + tid];
        sW3v[tid] = W3[n * HDIM + tid];
    }

    // ---- async DMA staging: sX (2 chunks/thread) + sW1t (4) first, then sW2t (32) ----
#pragma unroll
    for (int it = 0; it < 2; ++it) {            // 512 x 16B
        int q = tid + it * 256, r = q >> 2, c = q & 3;
        async_b128(sx_b + (unsigned)(r * S32 + c * 8) * 2,
                   (unsigned)(r * KP + c * 8) * 2, gx);
    }
#pragma unroll
    for (int it = 0; it < 4; ++it) {            // 1024 x 16B
        int q = tid + it * 256, r = q >> 2, c = q & 3;
        async_b128(sw1_b + (unsigned)(r * S32 + c * 8) * 2,
                   (unsigned)(r * KP + c * 8) * 2, gw1);
    }
#pragma unroll
    for (int it = 0; it < 32; ++it) {           // 8192 x 16B (W2^T, overlapped w/ layer 1)
        int q = tid + it * 256, r = q >> 5, c = q & 31;
        async_b128(sw2_b + (unsigned)(r * SH + c * 8) * 2,
                   (unsigned)(r * HDIM + c * 8) * 2, gw2);
    }
    asm volatile("s_wait_asynccnt 32" ::: "memory");  // in-order: first 6/thread (sX,W1t) done
    __syncthreads();

    // ---- layer 1: wave w owns row-stripe rT=w; A-frag kept in registers ----
    {
        bf16x16 a = load_frag(sX + wave * 16 * S32, S32, lane);
        const int colb = lane & 15, g = (lane >> 4) & 1;
#pragma unroll 4
        for (int cT = 0; cT < HDIM / 16; ++cT) {
            bf16x16 wfr = load_frag(sW1t + cT * 16 * S32, S32, lane);
            f32x8 c = {};
            c = __builtin_amdgcn_wmma_f32_16x16x32_bf16(false, a, false, wfr, (short)0, c, false, false);
            int col = cT * 16 + colb;
            float bv = sB1[col];
#pragma unroll
            for (int r = 0; r < 8; ++r) {
                int row = wave * 16 + r + 8 * g;
                sH1[row * SH + col] = f2bf_hw(fmaxf(c[r] + bv, 0.f));
            }
        }
    }
    asm volatile("s_wait_asynccnt 0" ::: "memory");   // W2^T DMA complete
    __syncthreads();

    // ---- layer 2: K=256; 8 A-frags (64 VGPRs) register-resident, reused over 16 cTs ----
    {
        bf16x16 a[HDIM / 32];
#pragma unroll
        for (int kc = 0; kc < HDIM / 32; ++kc)
            a[kc] = load_frag(sH1 + wave * 16 * SH + kc * 32, SH, lane);
        const int colb = lane & 15, g = (lane >> 4) & 1;
        for (int cT = 0; cT < HDIM / 16; ++cT) {
            f32x8 c = {};
#pragma unroll
            for (int kc = 0; kc < HDIM / 32; ++kc) {
                bf16x16 wfr = load_frag(sW2t + cT * 16 * SH + kc * 32, SH, lane);
                c = __builtin_amdgcn_wmma_f32_16x16x32_bf16(false, a[kc], false, wfr, (short)0, c, false, false);
            }
            int col = cT * 16 + colb;
            float bv = sB2[col];
#pragma unroll
            for (int r = 0; r < 8; ++r) {
                int row = wave * 16 + r + 8 * g;
                sH2[row * SH + col] = f2bf_hw(fmaxf(c[r] + bv, 0.f));
            }
        }
    }
    __syncthreads();

    // ---- layer 3: 256-wide dot per row (negligible FLOPs) ----
    if (tid < BT) {
        const unsigned short* hrow = sH2 + tid * SH;
        float acc = 0.f;
#pragma unroll 8
        for (int k = 0; k < HDIM; ++k) acc += bf2f(hrow[k]) * sW3v[k];
        out[(size_t)n * B_TOT + rowBase + tid] = acc + b3[n];
    }
}

extern "C" void kernel_launch(void* const* d_in, const int* in_sizes, int n_in,
                              void* d_out, int out_size, void* d_ws, size_t ws_size,
                              hipStream_t stream) {
    const float* state  = (const float*)d_in[0];
    const float* action = (const float*)d_in[1];
    const float* W1 = (const float*)d_in[2];
    const float* b1 = (const float*)d_in[3];
    const float* W2 = (const float*)d_in[4];
    const float* b2 = (const float*)d_in[5];
    const float* W3 = (const float*)d_in[6];
    const float* b3 = (const float*)d_in[7];
    float* out = (float*)d_out;
    unsigned short* ws = (unsigned short*)d_ws;   // needs ~3.6 MB

    const int total = WS_X_ELEMS + WS_W1_ELEMS + WS_W2_ELEMS;
    prep_bf16<<<(total + 255) / 256, 256, 0, stream>>>(state, action, W1, W2, ws);

    dim3 grid(B_TOT / BT, NQ);   // 256 x 10
    ensemble_q_fused<<<grid, 256, 0, stream>>>(ws, b1, b2, W3, b3, out);
}
